// JKNetConcat_17162689314846
// MI455X (gfx1250) — compile-verified, matrix operand order checked by reading, not run on hardware
//
#include <hip/hip_runtime.h>
#include <hip/hip_bf16.h>

typedef __attribute__((ext_vector_type(16))) __bf16 v16bf;
typedef __attribute__((ext_vector_type(8)))  float  v8f;
typedef __attribute__((ext_vector_type(4)))  unsigned int v4u;
typedef __attribute__((ext_vector_type(8)))  int v8i;
typedef __attribute__((ext_vector_type(4)))  int v4i;

#define N_NODES   50000
#define IN_F      128
#define UNITS     64
#define OUT_F     40
#define CAT_F     384             // 6 * 64
#define ROW_TILES (N_NODES / 16)  // 3125 (exact)
#define GEMM_BLOCKS 120

// round-to-nearest-even float -> bf16 bits (cold paths)
__device__ __forceinline__ unsigned short f2bf(float f) {
    unsigned int u = __float_as_uint(f);
    u += 0x7FFFu + ((u >> 16) & 1u);
    return (unsigned short)(u >> 16);
}

// K offset of packed pair j (0..7) within a 16x32 A / 32x16 B bf16 tile,
// before adding half*8 (ISA 7.12.2, 16-bit matrix tables)
__device__ __forceinline__ int kmap(int j) { return (j < 4) ? (j * 2) : (16 + (j - 4) * 2); }

union V16 { v16bf v; unsigned short u[16]; };

// ---------------------------------------------------------------------------
// Tensor Data Mover: DMA `bytes` (multiple of 8) from global to LDS offset.
// D# per CDNA5 ISA: group0 = {count=1 | lds_addr | global_addr |
// type=2}; group1 describes a 1-row tile of (bytes/8) 8-byte elements.
// Issued by one wave; completion via s_wait_tensorcnt.  (6-arg builtin form)
// ---------------------------------------------------------------------------
__device__ __forceinline__ void tdm_load_to_lds(const void* gptr, unsigned ldsOff, unsigned bytes) {
    unsigned long long ga = (unsigned long long)gptr;
    unsigned ne = bytes >> 3;                       // 8-byte elements
    v4u g0;
    g0.x = 1u;                                      // count = 1 (valid descriptor)
    g0.y = ldsOff;                                  // lds_addr  (bits 63:32)
    g0.z = (unsigned)(ga & 0xFFFFFFFFu);            // global_addr[31:0]
    g0.w = (unsigned)((ga >> 32) & 0x01FFFFFFu) | (2u << 30);  // addr[56:32] | type=2
    v8i g1;
    g1[0] = (int)(3u << 16);                        // workgroup_mask=0, data_size=3 (8B)
    g1[1] = (int)((ne & 0xFFFFu) << 16);            // tensor_dim0[15:0] @ bits 63:48
    g1[2] = (int)((ne >> 16) | (1u << 16));         // tensor_dim0[31:16] | tensor_dim1=1
    g1[3] = (int)((ne & 0xFFFFu) << 16);            // tensor_dim1[31:16]=0 | tile_dim0=ne
    g1[4] = 1;                                      // tile_dim1=1, tile_dim2=0
    g1[5] = (int)ne;                                // tensor_dim0_stride[31:0]
    g1[6] = 0;
    g1[7] = 0;
    v4i z4 = {0, 0, 0, 0};
    v8i z8 = {0, 0, 0, 0, 0, 0, 0, 0};
    __builtin_amdgcn_tensor_load_to_lds(g0, g1, z4, z4, z8, 0);
}

// ---------------------------------------------------------------------------
// Prepack: swizzle fp32 weights into WMMA bf16 B-operand layout.
// Per tile (kt,nt): 32 lanes x 16 bf16, lane l covers column n = nt*16+(l&15),
// packed K pairs at kt*32 + (l>>4)*8 + kmap(j) + {0,1}.
// ---------------------------------------------------------------------------
__device__ void pack_b(const float* W, int K, int stride, int nvalid,
                       int ktiles, int ntiles, unsigned short* out,
                       int tid, int nthr) {
    int total = ktiles * ntiles * 512;
    for (int t = tid; t < total; t += nthr) {
        int tile = t >> 9, e = t & 511;
        int kt = tile / ntiles, nt = tile - kt * ntiles;
        int lane = e >> 4, j2 = e & 15;
        int j = j2 >> 1, pos = j2 & 1;
        int n = nt * 16 + (lane & 15);
        int k = kt * 32 + (lane >> 4) * 8 + kmap(j) + pos;
        float v = (k < K && n < nvalid) ? W[k * stride + n] : 0.0f;
        out[t] = f2bf(v);
    }
}

__global__ void prepack_kernel(const float* w0_lin, const float* b0_lin,
                               const float* w0_self, const float* b0_self, const float* bias0,
                               const float* w_lin, const float* b_lin,
                               const float* w_self, const float* b_self, const float* bias,
                               const float* w_last, const float* b_last,
                               unsigned short* Bp0l, unsigned short* Bp0s,
                               unsigned short* Bpl, unsigned short* Bps,
                               unsigned short* Bplast, float* cvec, float* blast_pad) {
    int tid = threadIdx.x, nthr = blockDim.x;
    pack_b(w0_lin, 128, 64, 64, 4, 4, Bp0l, tid, nthr);
    pack_b(w0_self, 128, 64, 64, 4, 4, Bp0s, tid, nthr);
    for (int L = 0; L < 5; ++L) {
        pack_b(w_lin + L * 64 * 64, 64, 64, 64, 2, 4, Bpl + L * 4096, tid, nthr);
        pack_b(w_self + L * 64 * 64, 64, 64, 64, 2, 4, Bps + L * 4096, tid, nthr);
    }
    pack_b(w_last, 384, 40, 40, 12, 3, Bplast, tid, nthr);
    if (tid < 64) {
        cvec[tid] = b0_lin[tid] + b0_self[tid] + bias0[tid];
        for (int L = 0; L < 5; ++L)
            cvec[(L + 1) * 64 + tid] = b_lin[L * 64 + tid] + b_self[L * 64 + tid] + bias[L * 64 + tid];
    }
    if (tid < 48) blast_pad[tid] = (tid < 40) ? b_last[tid] : 0.0f;
}

// ---------------------------------------------------------------------------
// Dual GEMM: P = A@Wl and S = A@Ws with shared A operand.
// B: TDM -> LDS once per block -> hoisted into VGPRs once per wave, so the
// persistent row-tile loop is pure {A loads, cvt, WMMA stream, stores}.
// ---------------------------------------------------------------------------
template <int KTILES>
__global__ void __launch_bounds__(256)
gemm_dual(const float* __restrict__ A, int lda,
          const unsigned short* __restrict__ Bl, const unsigned short* __restrict__ Bs,
          float* __restrict__ P, float* __restrict__ S) {
    __shared__ unsigned short bsm[KTILES * 4 * 512 * 2];   // both B matrices
    constexpr int TB = KTILES * 4;                          // tiles per matrix
    constexpr unsigned bytesB = TB * 1024u;
    if (threadIdx.x < 32) {                                 // wave 0 drives the TDM
        unsigned base = (unsigned)(size_t)(&bsm[0]);
        tdm_load_to_lds(Bl, base, bytesB);
        tdm_load_to_lds(Bs, base + bytesB, bytesB);
        __builtin_amdgcn_s_wait_tensorcnt(0);
    }
    __syncthreads();

    const int lane = threadIdx.x & 31;
    const int half = lane >> 4;
    const int mn   = lane & 15;

    // hoist B into registers: invariant across all row tiles of this wave
    v16bf bl[TB], bs[TB];
#pragma unroll
    for (int t = 0; t < TB; ++t) {
        bl[t] = *(const v16bf*)(bsm + (size_t)t * 512 + lane * 16);
        bs[t] = *(const v16bf*)(bsm + (size_t)(TB + t) * 512 + lane * 16);
    }

    const int wpb  = blockDim.x >> 5;
    const int wstride = gridDim.x * wpb;

    for (int tile = blockIdx.x * wpb + (threadIdx.x >> 5); tile < ROW_TILES; tile += wstride) {
        const int m0 = tile * 16;
        v8f zero = {0, 0, 0, 0, 0, 0, 0, 0};
        v8f cp[4], cs[4];
#pragma unroll
        for (int nt = 0; nt < 4; ++nt) { cp[nt] = zero; cs[nt] = zero; }

#pragma unroll
        for (int kt = 0; kt < KTILES; ++kt) {
            V16 a;
            const float* Arow = A + (size_t)(m0 + mn) * lda + kt * 32 + half * 8;
#pragma unroll
            for (int j = 0; j < 8; ++j) {
                float2 f = *(const float2*)(Arow + kmap(j));
                a.v[2 * j]     = (__bf16)f.x;    // native v_cvt, RNE
                a.v[2 * j + 1] = (__bf16)f.y;
            }
#pragma unroll
            for (int nt = 0; nt < 4; ++nt) {
                cp[nt] = __builtin_amdgcn_wmma_f32_16x16x32_bf16(false, a.v, false, bl[kt * 4 + nt], (short)0, cp[nt], false, false);
                cs[nt] = __builtin_amdgcn_wmma_f32_16x16x32_bf16(false, a.v, false, bs[kt * 4 + nt], (short)0, cs[nt], false, false);
            }
        }
#pragma unroll
        for (int nt = 0; nt < 4; ++nt) {
#pragma unroll
            for (int r = 0; r < 8; ++r) {
                int m = m0 + half * 8 + r;
                int n = nt * 16 + mn;
                P[(size_t)m * 64 + n] = cp[nt][r];
                S[(size_t)m * 64 + n] = cs[nt][r];
            }
        }
    }
}

// ---------------------------------------------------------------------------
// Edge aggregation: agg[dst] += p[src]   (16 threads / edge, float4 gathers,
// global_atomic_add_f32 no-return scatter; node features are L2-resident)
// ---------------------------------------------------------------------------
__global__ void __launch_bounds__(256)
aggregate_kernel(const float* __restrict__ p, const int* __restrict__ src,
                 const int* __restrict__ dst, float* __restrict__ agg, int nEdges) {
    int t = blockIdx.x * blockDim.x + threadIdx.x;
    int e = t >> 4;
    if (e >= nEdges) return;
    int c = (t & 15) << 2;
    int sN = src[e], dN = dst[e];
    const float4 v = *(const float4*)(p + (size_t)sN * 64 + c);
    float* o = agg + (size_t)dN * 64 + c;
    atomicAdd(o + 0, v.x);
    atomicAdd(o + 1, v.y);
    atomicAdd(o + 2, v.z);
    atomicAdd(o + 3, v.w);
}

__global__ void fill_zero(float* __restrict__ p, int n) {
    int i = blockIdx.x * blockDim.x + threadIdx.x;
    if (i < n) p[i] = 0.0f;
}

// h = relu(agg + s + cvec[layer]); write bf16 slice into hcat; re-zero agg
// so the next layer's aggregation starts from zero (saves a fill pass).
__global__ void __launch_bounds__(256)
combine_kernel(float* __restrict__ agg, const float* __restrict__ s,
               const float* __restrict__ cvec, float* __restrict__ h,
               unsigned short* __restrict__ hcat, int layer) {
    int i = blockIdx.x * blockDim.x + threadIdx.x;
    if (i >= N_NODES * UNITS) return;
    int row = i >> 6, j = i & 63;
    float v = agg[i] + s[i] + cvec[layer * 64 + j];
    agg[i] = 0.0f;
    v = v > 0.0f ? v : 0.0f;
    h[i] = v;
    hcat[(size_t)row * CAT_F + layer * 64 + j] = f2bf(v);
}

// ---------------------------------------------------------------------------
// Final GEMM: out[50000,40] = hcat(bf16)[50000,384] @ w_last + b_last
// K = 12 tiles of 32, N padded 40 -> 48 (3 tiles of 16).
// B: TDM -> LDS -> registers (36 tiles, invariant over row tiles).
// ---------------------------------------------------------------------------
__global__ void __launch_bounds__(256)
gemm_final(const unsigned short* __restrict__ Acat,
           const unsigned short* __restrict__ Bp,
           const float* __restrict__ bias, float* __restrict__ out) {
    __shared__ unsigned short bsm[18432];           // 36 KB: 12kt x 3nt tiles
    if (threadIdx.x < 32) {
        tdm_load_to_lds(Bp, (unsigned)(size_t)(&bsm[0]), 36u * 1024u);
        __builtin_amdgcn_s_wait_tensorcnt(0);
    }
    __syncthreads();

    const int lane = threadIdx.x & 31;
    const int half = lane >> 4;
    const int mn   = lane & 15;

    v16bf b[36];
#pragma unroll
    for (int t = 0; t < 36; ++t)
        b[t] = *(const v16bf*)(bsm + (size_t)t * 512 + lane * 16);

    const int wpb  = blockDim.x >> 5;
    const int wstride = gridDim.x * wpb;

    for (int tile = blockIdx.x * wpb + (threadIdx.x >> 5); tile < ROW_TILES; tile += wstride) {
        const int m0 = tile * 16;
        v8f zero = {0, 0, 0, 0, 0, 0, 0, 0};
        v8f c[3];
#pragma unroll
        for (int nt = 0; nt < 3; ++nt) c[nt] = zero;

#pragma unroll
        for (int kt = 0; kt < 12; ++kt) {
            V16 a;
            const unsigned short* Arow = Acat + (size_t)(m0 + mn) * CAT_F + kt * 32 + half * 8;
#pragma unroll
            for (int j = 0; j < 8; ++j) {
                unsigned int w = *(const unsigned int*)(Arow + kmap(j));  // 4B-aligned pair
                a.u[2 * j]     = (unsigned short)(w & 0xFFFFu);
                a.u[2 * j + 1] = (unsigned short)(w >> 16);
            }
#pragma unroll
            for (int nt = 0; nt < 3; ++nt)
                c[nt] = __builtin_amdgcn_wmma_f32_16x16x32_bf16(false, a.v, false, b[kt * 3 + nt], (short)0, c[nt], false, false);
        }
#pragma unroll
        for (int nt = 0; nt < 3; ++nt) {
#pragma unroll
            for (int r = 0; r < 8; ++r) {
                int n = nt * 16 + mn;
                if (n < OUT_F) {
                    int m = m0 + half * 8 + r;
                    out[(size_t)m * OUT_F + n] = c[nt][r] + bias[n];
                }
            }
        }
    }
}

// ---------------------------------------------------------------------------
extern "C" void kernel_launch(void* const* d_in, const int* in_sizes, int n_in,
                              void* d_out, int out_size, void* d_ws, size_t ws_size,
                              hipStream_t stream) {
    const float* x       = (const float*)d_in[0];
    const int*   src     = (const int*)d_in[1];
    const int*   dst     = (const int*)d_in[2];
    const float* w0_lin  = (const float*)d_in[3];
    const float* b0_lin  = (const float*)d_in[4];
    const float* w0_self = (const float*)d_in[5];
    const float* b0_self = (const float*)d_in[6];
    const float* bias0   = (const float*)d_in[7];
    const float* w_lin   = (const float*)d_in[8];
    const float* b_lin   = (const float*)d_in[9];
    const float* w_self  = (const float*)d_in[10];
    const float* b_self  = (const float*)d_in[11];
    const float* bias    = (const float*)d_in[12];
    const float* w_last  = (const float*)d_in[13];
    const float* b_last  = (const float*)d_in[14];
    float* out = (float*)d_out;
    const int nEdges = in_sizes[1];

    // workspace carve-out (256B aligned regions)
    char* ws = (char*)d_ws;
    size_t off = 0;
    auto carve = [&](size_t bytes) -> void* {
        void* ptr = ws + off;
        off = (off + bytes + 255) & ~(size_t)255;
        return ptr;
    };
    float* h   = (float*)carve((size_t)N_NODES * 64 * 4);
    float* p   = (float*)carve((size_t)N_NODES * 64 * 4);
    float* s   = (float*)carve((size_t)N_NODES * 64 * 4);
    float* agg = (float*)carve((size_t)N_NODES * 64 * 4);
    unsigned short* hcat   = (unsigned short*)carve((size_t)N_NODES * CAT_F * 2);
    unsigned short* Bp0l   = (unsigned short*)carve(16 * 512 * 2);      // 4kt x 4nt
    unsigned short* Bp0s   = (unsigned short*)carve(16 * 512 * 2);
    unsigned short* Bpl    = (unsigned short*)carve(5 * 8 * 512 * 2);   // 5 layers x (2kt x 4nt)
    unsigned short* Bps    = (unsigned short*)carve(5 * 8 * 512 * 2);
    unsigned short* Bplast = (unsigned short*)carve(36 * 512 * 2);      // 12kt x 3nt
    float* cvec      = (float*)carve(6 * 64 * 4);
    float* blast_pad = (float*)carve(48 * 4);

    prepack_kernel<<<1, 256, 0, stream>>>(w0_lin, b0_lin, w0_self, b0_self, bias0,
                                          w_lin, b_lin, w_self, b_self, bias,
                                          w_last, b_last,
                                          Bp0l, Bp0s, Bpl, Bps, Bplast, cvec, blast_pad);

    const int elemBlocks = (N_NODES * 64 + 255) / 256;
    const int aggBlocks  = (nEdges * 16 + 255) / 256;

    // agg starts from zero once; combine_kernel re-zeroes it for the next layer
    fill_zero<<<elemBlocks, 256, 0, stream>>>(agg, N_NODES * 64);

    // ---- layer 0 (project 128->64 first, then aggregate: linearity) ----
    gemm_dual<4><<<GEMM_BLOCKS, 256, 0, stream>>>(x, IN_F, Bp0l, Bp0s, p, s);
    aggregate_kernel<<<aggBlocks, 256, 0, stream>>>(p, src, dst, agg, nEdges);
    combine_kernel<<<elemBlocks, 256, 0, stream>>>(agg, s, cvec, h, hcat, 0);

    // ---- layers 1..5 ----
    for (int L = 0; L < 5; ++L) {
        gemm_dual<2><<<GEMM_BLOCKS, 256, 0, stream>>>(h, UNITS, Bpl + L * 4096, Bps + L * 4096, p, s);
        aggregate_kernel<<<aggBlocks, 256, 0, stream>>>(p, src, dst, agg, nEdges);
        combine_kernel<<<elemBlocks, 256, 0, stream>>>(agg, s, cvec, h, hcat, L + 1);
    }

    // ---- jumping-knowledge output GEMM ----
    gemm_final<<<GEMM_BLOCKS, 256, 0, stream>>>(hcat, Bplast, blast_pad, out);
}
